// MultiHeadAttention_42545946034655
// MI455X (gfx1250) — matrix operand order repaired and checked by reading, not compiled, not run on hardware
//
#include <hip/hip_runtime.h>
#include <hip/hip_bf16.h>
#include <stdint.h>

typedef __attribute__((ext_vector_type(16))) __bf16 v16bf;
typedef __attribute__((ext_vector_type(8)))  __bf16 v8bf;
typedef __attribute__((ext_vector_type(8)))  float  v8f;

#define DEV static __device__ __forceinline__

// ---------------- CDNA5 async copy (global -> LDS), ASYNCcnt-tracked ----------------
DEV unsigned lds_off_u32(const void* p) {
  // LDS aperture: flat addr[31:0] == LDS byte offset
  return (unsigned)(uintptr_t)p;
}
DEV void async_ld_b128(unsigned lds_off, const void* gaddr) {
  asm volatile("global_load_async_to_lds_b128 %0, %1, off"
               :: "v"(lds_off), "v"(gaddr) : "memory");
}
DEV void wait_async0() { asm volatile("s_wait_asynccnt 0x0" ::: "memory"); }

// ---------------- WMMA fragment helpers (bf16, 16x16x32) ----------------
// A-matrix 16x32 (MxK) ISA layout:
//   lanes 0-15 : elems 0-7 -> K=0..7,  elems 8-15 -> K=16..23  (row M = lane)
//   lanes 16-31: elems 0-7 -> K=8..15, elems 8-15 -> K=24..31  (row M = lane-16)
DEV v16bf frag_a_lds(const __bf16* base, int ldk, int lane) {
  const int half = lane >> 4, row = lane & 15;
  const __bf16* p = base + row * ldk + half * 8;
  v8bf lo = *(const v8bf*)(p);
  v8bf hi = *(const v8bf*)(p + 16);
  v16bf r;
#pragma unroll
  for (int i = 0; i < 8; ++i) { r[i] = lo[i]; r[i + 8] = hi[i]; }
  return r;
}
// B-matrix 32x16 (KxN) staged N-major [N][K] in LDS:
//   lanes 0-15: K=0..7,16..23, lanes 16-31: K=8..15,24..31 (col N = lane&15)
DEV v16bf frag_b_lds(const __bf16* base, int ldk, int lane) {
  const int half = lane >> 4, n = lane & 15;
  const __bf16* p = base + n * ldk + half * 8;
  v8bf lo = *(const v8bf*)(p);
  v8bf hi = *(const v8bf*)(p + 16);
  v16bf r;
#pragma unroll
  for (int i = 0; i < 8; ++i) { r[i] = lo[i]; r[i + 8] = hi[i]; }
  return r;
}
DEV v8f wmma_bf16(v16bf a, v16bf b, v8f c) {
  return __builtin_amdgcn_wmma_f32_16x16x32_bf16(false, a, false, b, (short)0, c,
                                                 false, false);
}

// ---------------- constants ----------------
static constexpr int B = 4, S = 2048, DM = 768, NH = 12, DA = 64;
static constexpr int BH = B * NH;
static constexpr int LDK = 72;                       // pitch: 144B, 16B aligned
static constexpr int WN  = 4 * DM * DM;              // weight elems
static constexpr int IN4 = 3 * (B * S) * DM / 4;     // input float4 groups

// ---------------- kernel 1: convert weights (transpose) + inputs to bf16 ----------
__global__ __launch_bounds__(256)
void mha_convert(const float* __restrict__ q, const float* __restrict__ k,
                 const float* __restrict__ v,
                 const float* __restrict__ Wq, const float* __restrict__ Wk,
                 const float* __restrict__ Wv, const float* __restrict__ Wo,
                 __bf16* __restrict__ Wqkv_t, __bf16* __restrict__ Wo_t,
                 __bf16* __restrict__ Xq, __bf16* __restrict__ Xk,
                 __bf16* __restrict__ Xv) {
  int idx = blockIdx.x * 256 + threadIdx.x;
  if (idx < WN) {                       // W[k][n] -> Wt[n][k]  (coalesced read)
    int r = idx % DM, c = (idx / DM) % DM, t = idx / (DM * DM);
    const float* W = (t == 0) ? Wq : (t == 1) ? Wk : (t == 2) ? Wv : Wo;
    __bf16 val = (__bf16)W[(size_t)c * DM + r];
    if (t < 3) Wqkv_t[((size_t)t * DM + r) * DM + c] = val;
    else       Wo_t[(size_t)r * DM + c] = val;
  } else if (idx < WN + IN4) {          // straight f32 -> bf16, 4 at a time
    int id = idx - WN;
    int t = id / (IN4 / 3), e = id % (IN4 / 3);
    const float* X = (t == 0) ? q : (t == 1) ? k : v;
    __bf16* D = ((t == 0) ? Xq : (t == 1) ? Xk : Xv) + (size_t)e * 4;
    float4 xv = ((const float4*)X)[e];
    D[0] = (__bf16)xv.x; D[1] = (__bf16)xv.y; D[2] = (__bf16)xv.z; D[3] = (__bf16)xv.w;
  }
}

// ---------------- kernel 2: fused QKV projection GEMM (async double-buffered) -----
// 128 thr = 4 waves; block tile 64x128, wave tile 32x64; BK=64.
__global__ __launch_bounds__(128)
void mha_gemm_qkv(const __bf16* __restrict__ Xq, const __bf16* __restrict__ Xk,
                  const __bf16* __restrict__ Xv, const __bf16* __restrict__ Wt,
                  const float* __restrict__ bq, const float* __restrict__ bk,
                  const float* __restrict__ bv,
                  __bf16* __restrict__ Qh, __bf16* __restrict__ Kh,
                  __bf16* __restrict__ Vt) {
  __shared__ __attribute__((aligned(16))) __bf16 As[2][64 * LDK];
  __shared__ __attribute__((aligned(16))) __bf16 Bs[2][128 * LDK];
  const int tid = threadIdx.x, lane = tid & 31, wave = tid >> 5;
  const int m0 = blockIdx.x * 64, n0 = blockIdx.y * 128;
  const int t = blockIdx.y / 6;                       // 0=Q 1=K 2=V
  const __bf16* X    = (t == 0) ? Xq : (t == 1) ? Xk : Xv;
  const float*  bias = (t == 0) ? bq : (t == 1) ? bk : bv;
  const int wm = wave >> 1, wn = wave & 1;            // 2x2 waves -> 32x64 each

  auto stage = [&](int buf, int k0) {
#pragma unroll
    for (int i = 0; i < 4; ++i) {                     // A: 64x64 bf16
      int idx = i * 128 + tid, r = idx >> 3, c8 = idx & 7;
      async_ld_b128(lds_off_u32(&As[buf][r * LDK + c8 * 8]),
                    X + (size_t)(m0 + r) * DM + k0 + c8 * 8);
    }
#pragma unroll
    for (int i = 0; i < 8; ++i) {                     // B: 128x64 bf16
      int idx = i * 128 + tid, r = idx >> 3, c8 = idx & 7;
      async_ld_b128(lds_off_u32(&Bs[buf][r * LDK + c8 * 8]),
                    Wt + (size_t)(n0 + r) * DM + k0 + c8 * 8);
    }
  };

  stage(0, 0);
  v8f acc[2][4] = {};
  for (int kt = 0; kt < DM / 64; ++kt) {
    wait_async0();
    __syncthreads();
    if (kt + 1 < DM / 64) stage((kt + 1) & 1, (kt + 1) * 64);
    const __bf16* Ab = As[kt & 1];
    const __bf16* Bb = Bs[kt & 1];
#pragma unroll
    for (int ks = 0; ks < 2; ++ks) {
      v16bf a0 = frag_a_lds(Ab + (wm * 32) * LDK + ks * 32, LDK, lane);
      v16bf a1 = frag_a_lds(Ab + (wm * 32 + 16) * LDK + ks * 32, LDK, lane);
#pragma unroll
      for (int ni = 0; ni < 4; ++ni) {
        v16bf b = frag_b_lds(Bb + (wn * 64 + ni * 16) * LDK + ks * 32, LDK, lane);
        acc[0][ni] = wmma_bf16(a0, b, acc[0][ni]);
        acc[1][ni] = wmma_bf16(a1, b, acc[1][ni]);
      }
    }
  }
  // epilogue: Q scaled 1/8 [bh][s][64]; K [bh][s][64]; V transposed [bh][64][s]
  const int half = lane >> 4, l15 = lane & 15;
#pragma unroll
  for (int mi = 0; mi < 2; ++mi)
#pragma unroll
    for (int ni = 0; ni < 4; ++ni) {
      int nl = (n0 - t * DM) + wn * 64 + ni * 16 + l15;
      float bb = bias[nl];
      int h = nl >> 6, d = nl & 63;
#pragma unroll
      for (int j = 0; j < 8; ++j) {
        int m = m0 + wm * 32 + mi * 16 + half * 8 + j;
        int bi = m >> 11, s = m & (S - 1);
        float val = acc[mi][ni][j] + bb;
        size_t bh = (size_t)bi * NH + h;
        if (t == 0)      Qh[(bh * S + s) * DA + d] = (__bf16)(val * 0.125f);
        else if (t == 1) Kh[(bh * S + s) * DA + d] = (__bf16)val;
        else             Vt[(bh * DA + d) * S + s] = (__bf16)val;
      }
    }
}

// ---------------- kernel 3: flash attention (async double-buffered K/V) ----------
__global__ __launch_bounds__(128)
void mha_attention(const __bf16* __restrict__ Qh, const __bf16* __restrict__ Kh,
                   const __bf16* __restrict__ Vt, __bf16* __restrict__ AOut) {
  __shared__ __attribute__((aligned(16))) __bf16 Qs[64 * LDK];
  __shared__ __attribute__((aligned(16))) __bf16 Ks[2][64 * LDK];
  __shared__ __attribute__((aligned(16))) __bf16 Vs[2][64 * LDK];    // [d][kv]
  __shared__ __attribute__((aligned(16))) __bf16 Ps[4][16 * LDK];
  const int tid = threadIdx.x, lane = tid & 31, wave = tid >> 5;
  const int bh = blockIdx.x, q0 = blockIdx.y * 64;
  const size_t qkb = (size_t)bh * S * DA;
  const __bf16* Vb = Vt + (size_t)bh * DA * S;
  const int half = lane >> 4, l15 = lane & 15;

  auto stage_kv = [&](int buf, int kv0) {
#pragma unroll
    for (int i = 0; i < 4; ++i) {
      int idx = i * 128 + tid, r = idx >> 3, c8 = idx & 7;
      async_ld_b128(lds_off_u32(&Ks[buf][r * LDK + c8 * 8]),
                    Kh + qkb + (size_t)(kv0 + r) * DA + c8 * 8);
      async_ld_b128(lds_off_u32(&Vs[buf][r * LDK + c8 * 8]),
                    Vb + (size_t)r * S + kv0 + c8 * 8);
    }
  };
#pragma unroll
  for (int i = 0; i < 4; ++i) {                        // stage Q once (async)
    int idx = i * 128 + tid, r = idx >> 3, c8 = idx & 7;
    async_ld_b128(lds_off_u32(&Qs[r * LDK + c8 * 8]),
                  Qh + qkb + (size_t)(q0 + r) * DA + c8 * 8);
  }
  stage_kv(0, 0);

  float m_r[8], l_r[8];
  v8f o[4] = {};
#pragma unroll
  for (int j = 0; j < 8; ++j) { m_r[j] = -1e30f; l_r[j] = 0.f; }

  for (int it = 0; it < S / 64; ++it) {
    wait_async0();
    __syncthreads();
    if (it + 1 < S / 64) stage_kv((it + 1) & 1, (it + 1) * 64);
    const __bf16* Kb = Ks[it & 1];
    const __bf16* Vl = Vs[it & 1];

    v8f sacc[4] = {};
#pragma unroll
    for (int ks = 0; ks < 2; ++ks) {
      v16bf aq = frag_a_lds(Qs + (wave * 16) * LDK + ks * 32, LDK, lane);
#pragma unroll
      for (int nt = 0; nt < 4; ++nt) {
        v16bf bk_ = frag_b_lds(Kb + (nt * 16) * LDK + ks * 32, LDK, lane);
        sacc[nt] = wmma_bf16(aq, bk_, sacc[nt]);
      }
    }
    float mnew[8], fac[8];
#pragma unroll
    for (int j = 0; j < 8; ++j) {
      float mx = fmaxf(fmaxf(sacc[0][j], sacc[1][j]), fmaxf(sacc[2][j], sacc[3][j]));
      mx = fmaxf(mx, __shfl_xor(mx, 1, 32));
      mx = fmaxf(mx, __shfl_xor(mx, 2, 32));
      mx = fmaxf(mx, __shfl_xor(mx, 4, 32));
      mx = fmaxf(mx, __shfl_xor(mx, 8, 32));
      mnew[j] = fmaxf(m_r[j], mx);
      fac[j] = __expf(m_r[j] - mnew[j]);
      m_r[j] = mnew[j];
    }
#pragma unroll
    for (int nt = 0; nt < 4; ++nt)
#pragma unroll
      for (int j = 0; j < 8; ++j)
        sacc[nt][j] = __expf(sacc[nt][j] - mnew[j]);
#pragma unroll
    for (int j = 0; j < 8; ++j) {
      float sm = sacc[0][j] + sacc[1][j] + sacc[2][j] + sacc[3][j];
      sm += __shfl_xor(sm, 1, 32);
      sm += __shfl_xor(sm, 2, 32);
      sm += __shfl_xor(sm, 4, 32);
      sm += __shfl_xor(sm, 8, 32);
      l_r[j] = l_r[j] * fac[j] + sm;
    }
#pragma unroll
    for (int nt = 0; nt < 4; ++nt)
#pragma unroll
      for (int j = 0; j < 8; ++j)
        o[nt][j] *= fac[j];
    __bf16* pw = Ps[wave];
#pragma unroll
    for (int nt = 0; nt < 4; ++nt)
#pragma unroll
      for (int j = 0; j < 8; ++j)
        pw[(half * 8 + j) * LDK + nt * 16 + l15] = (__bf16)sacc[nt][j];
#pragma unroll
    for (int ks = 0; ks < 2; ++ks) {
      v16bf ap = frag_a_lds(pw + ks * 32, LDK, lane);
#pragma unroll
      for (int nt = 0; nt < 4; ++nt) {
        v16bf bv_ = frag_b_lds(Vl + (nt * 16) * LDK + ks * 32, LDK, lane);
        o[nt] = wmma_bf16(ap, bv_, o[nt]);
      }
    }
  }
  const int b = bh / NH, h = bh % NH;
#pragma unroll
  for (int j = 0; j < 8; ++j) {
    float inv = 1.0f / l_r[j];
    int s = q0 + wave * 16 + half * 8 + j;
    size_t rowb = ((size_t)b * S + s) * DM + h * DA;
#pragma unroll
    for (int nt = 0; nt < 4; ++nt)
      AOut[rowb + nt * 16 + l15] = (__bf16)(o[nt][j] * inv);
  }
}

// ---------------- kernel 4: output projection (async double-buffered) -----------
__global__ __launch_bounds__(128)
void mha_gemm_out(const __bf16* __restrict__ A, const __bf16* __restrict__ Wt,
                  const float* __restrict__ bo, float* __restrict__ out) {
  __shared__ __attribute__((aligned(16))) __bf16 As[2][64 * LDK];
  __shared__ __attribute__((aligned(16))) __bf16 Bs[2][128 * LDK];
  const int tid = threadIdx.x, lane = tid & 31, wave = tid >> 5;
  const int m0 = blockIdx.x * 64, n0 = blockIdx.y * 128;
  const int wm = wave >> 1, wn = wave & 1;

  auto stage = [&](int buf, int k0) {
#pragma unroll
    for (int i = 0; i < 4; ++i) {
      int idx = i * 128 + tid, r = idx >> 3, c8 = idx & 7;
      async_ld_b128(lds_off_u32(&As[buf][r * LDK + c8 * 8]),
                    A + (size_t)(m0 + r) * DM + k0 + c8 * 8);
    }
#pragma unroll
    for (int i = 0; i < 8; ++i) {
      int idx = i * 128 + tid, r = idx >> 3, c8 = idx & 7;
      async_ld_b128(lds_off_u32(&Bs[buf][r * LDK + c8 * 8]),
                    Wt + (size_t)(n0 + r) * DM + k0 + c8 * 8);
    }
  };

  stage(0, 0);
  v8f acc[2][4] = {};
  for (int kt = 0; kt < DM / 64; ++kt) {
    wait_async0();
    __syncthreads();
    if (kt + 1 < DM / 64) stage((kt + 1) & 1, (kt + 1) * 64);
    const __bf16* Ab = As[kt & 1];
    const __bf16* Bb = Bs[kt & 1];
#pragma unroll
    for (int ks = 0; ks < 2; ++ks) {
      v16bf a0 = frag_a_lds(Ab + (wm * 32) * LDK + ks * 32, LDK, lane);
      v16bf a1 = frag_a_lds(Ab + (wm * 32 + 16) * LDK + ks * 32, LDK, lane);
#pragma unroll
      for (int ni = 0; ni < 4; ++ni) {
        v16bf b = frag_b_lds(Bb + (wn * 64 + ni * 16) * LDK + ks * 32, LDK, lane);
        acc[0][ni] = wmma_bf16(a0, b, acc[0][ni]);
        acc[1][ni] = wmma_bf16(a1, b, acc[1][ni]);
      }
    }
  }
  const int half = lane >> 4, l15 = lane & 15;
#pragma unroll
  for (int mi = 0; mi < 2; ++mi)
#pragma unroll
    for (int ni = 0; ni < 4; ++ni) {
      int n = n0 + wn * 64 + ni * 16 + l15;
      float bb = bo[n];
#pragma unroll
      for (int j = 0; j < 8; ++j) {
        int m = m0 + wm * 32 + mi * 16 + half * 8 + j;
        out[(size_t)m * DM + n] = acc[mi][ni][j] + bb;
      }
    }
}

// ---------------- host launch ----------------
extern "C" void kernel_launch(void* const* d_in, const int* in_sizes, int n_in,
                              void* d_out, int out_size, void* d_ws, size_t ws_size,
                              hipStream_t stream) {
  const float* q  = (const float*)d_in[0];
  const float* k  = (const float*)d_in[1];
  const float* v  = (const float*)d_in[2];
  const float* Wq = (const float*)d_in[3];
  const float* bq = (const float*)d_in[4];
  const float* Wk = (const float*)d_in[5];
  const float* bk = (const float*)d_in[6];
  const float* Wv = (const float*)d_in[7];
  const float* bv = (const float*)d_in[8];
  const float* Wo = (const float*)d_in[9];
  const float* bo = (const float*)d_in[10];
  float* out = (float*)d_out;

  char* ws = (char*)d_ws;
  const size_t szW3 = (size_t)3 * DM * DM * sizeof(__bf16);
  const size_t szW1 = (size_t)DM * DM * sizeof(__bf16);
  const size_t szX  = (size_t)B * S * DM * sizeof(__bf16);   // 12.6 MB each
  const size_t szT  = (size_t)BH * S * DA * sizeof(__bf16);
  size_t off = 0;
  __bf16* Wqkv_t = (__bf16*)(ws + off); off += szW3;
  __bf16* Wo_t   = (__bf16*)(ws + off); off += szW1;
  __bf16* Xq     = (__bf16*)(ws + off); off += szX;
  __bf16* Xk     = (__bf16*)(ws + off); off += szX;
  __bf16* Xv     = (__bf16*)(ws + off); off += szX;
  __bf16* Qh     = (__bf16*)(ws + off); off += szT;
  __bf16* Kh     = (__bf16*)(ws + off); off += szT;
  __bf16* Vt     = (__bf16*)(ws + off); off += szT;
  __bf16* AOut   = (__bf16*)(ws + off); off += szX;
  (void)in_sizes; (void)n_in; (void)out_size; (void)ws_size;

  hipLaunchKernelGGL(mha_convert, dim3((WN + IN4 + 255) / 256), dim3(256), 0, stream,
                     q, k, v, Wq, Wk, Wv, Wo, Wqkv_t, Wo_t, Xq, Xk, Xv);
  hipLaunchKernelGGL(mha_gemm_qkv, dim3((B * S) / 64, (3 * DM) / 128), dim3(128),
                     0, stream, Xq, Xk, Xv, Wqkv_t, bq, bk, bv, Qh, Kh, Vt);
  hipLaunchKernelGGL(mha_attention, dim3(BH, S / 64), dim3(128),
                     0, stream, Qh, Kh, Vt, AOut);
  hipLaunchKernelGGL(mha_gemm_out, dim3((B * S) / 64, DM / 128), dim3(128),
                     0, stream, AOut, Wo_t, bo, out);
}
